// Attention_45037027066352
// MI455X (gfx1250) — compile-verified, matrix operand order checked by reading, not run on hardware
//
#include <hip/hip_runtime.h>

typedef __attribute__((ext_vector_type(16))) _Float16 v16h;
typedef __attribute__((ext_vector_type(8)))  _Float16 v8h;
typedef __attribute__((ext_vector_type(8)))  float    v8f;

#define WMMA_F32_F16(a, b, c) \
  __builtin_amdgcn_wmma_f32_16x16x32_f16(false, (a), false, (b), (short)0, (c), false, false)

// Load a 16-half WMMA fragment from a row-major matrix.
// A fragment: row m = base+ (lane&15), halves at k = kbase + (lane>>4)*8 + {0..7}
// and k = kbase + 16 + (lane>>4)*8 + {0..7}  (per CDNA5 16-bit A layout).
// The B fragment uses the identical pattern applied to B^T stored row-major [N,K].
__device__ __forceinline__ v16h load_frag(const _Float16* __restrict__ p, int ld,
                                          int row, int kbase) {
  const int lane = threadIdx.x & 31;
  const _Float16* q = p + (size_t)(row + (lane & 15)) * ld + (kbase + ((lane >> 4) << 3));
  v8h lo = *(const v8h*)(q);
  v8h hi = *(const v8h*)(q + 16);
  v16h r;
#pragma unroll
  for (int i = 0; i < 8; ++i) { r[i] = lo[i]; r[i + 8] = hi[i]; }
  return r;
}

// 32x32 output tile per wave (2x2 subtiles of 16x16), K-loop of 32.
__device__ __forceinline__ void gemm_core_32x32(const _Float16* __restrict__ A,
                                                const _Float16* __restrict__ Bt,
                                                int K, int m0, int n0, v8f c[2][2]) {
  for (int kb = 0; kb < K; kb += 32) {
    v16h a0 = load_frag(A, K, m0, kb);
    v16h a1 = load_frag(A, K, m0 + 16, kb);
    v16h b0 = load_frag(Bt, K, n0, kb);
    v16h b1 = load_frag(Bt, K, n0 + 16, kb);
    c[0][0] = WMMA_F32_F16(a0, b0, c[0][0]);
    c[0][1] = WMMA_F32_F16(a0, b1, c[0][1]);
    c[1][0] = WMMA_F32_F16(a1, b0, c[1][0]);
    c[1][1] = WMMA_F32_F16(a1, b1, c[1][1]);
  }
}

__global__ __launch_bounds__(256) void k_f32_to_f16(const float* __restrict__ s,
                                                    _Float16* __restrict__ d, int n) {
  int i = blockIdx.x * 256 + threadIdx.x;
  if (i < n) d[i] = (_Float16)s[i];
}

// qkv = x @ Wqkv^T with fused epilogue: relu; v *= 4; scatter per-head.
// Qr,Kr: [B,H,N,64] f16 ; Vt: [B,H,64,N] f16 (transposed for the P@V B-fragments).
__global__ __launch_bounds__(128) void k_gemm_qkv(const _Float16* __restrict__ A,
    const _Float16* __restrict__ Bt, int K,
    _Float16* __restrict__ Qr, _Float16* __restrict__ Kr, _Float16* __restrict__ Vt) {
  const int wave = threadIdx.x >> 5, lane = threadIdx.x & 31;
  const int m0 = blockIdx.y * 128 + wave * 32;
  const int n0 = blockIdx.x * 32;
  v8f c[2][2] = {};
  gemm_core_32x32(A, Bt, K, m0, n0, c);
  const int rbase = (lane >> 4) << 3, cl = lane & 15;
#pragma unroll
  for (int mi = 0; mi < 2; ++mi)
#pragma unroll
    for (int ni = 0; ni < 2; ++ni)
#pragma unroll
      for (int r = 0; r < 8; ++r) {
        const int row = m0 + mi * 16 + rbase + r;       // 0..8191  (b*1024+n)
        const int col = n0 + ni * 16 + cl;              // 0..2303  (sec*768+h*64+d)
        const float v = fmaxf(c[mi][ni][r], 0.0f);
        const int b = row >> 10, nidx = row & 1023;
        const int sec = col / 768, rem = col % 768;
        const int h = rem >> 6, d = rem & 63;
        const size_t bh = (size_t)(b * 12 + h);
        if (sec == 0)      Qr[(bh * 1024 + nidx) * 64 + d] = (_Float16)v;
        else if (sec == 1) Kr[(bh * 1024 + nidx) * 64 + d] = (_Float16)v;
        else               Vt[(bh * 64 + d) * 1024 + nidx] = (_Float16)(4.0f * v);
      }
}

__global__ __launch_bounds__(128) void k_gemm_bias_f16(const _Float16* __restrict__ A,
    const _Float16* __restrict__ Bt, const float* __restrict__ bias,
    _Float16* __restrict__ C, int N, int K) {
  const int wave = threadIdx.x >> 5, lane = threadIdx.x & 31;
  const int m0 = blockIdx.y * 128 + wave * 32;
  const int n0 = blockIdx.x * 32;
  v8f c[2][2] = {};
  gemm_core_32x32(A, Bt, K, m0, n0, c);
  const int rbase = (lane >> 4) << 3, cl = lane & 15;
#pragma unroll
  for (int mi = 0; mi < 2; ++mi)
#pragma unroll
    for (int ni = 0; ni < 2; ++ni)
#pragma unroll
      for (int r = 0; r < 8; ++r) {
        const int row = m0 + mi * 16 + rbase + r;
        const int col = n0 + ni * 16 + cl;
        C[(size_t)row * N + col] = (_Float16)(c[mi][ni][r] + bias[col]);
      }
}

__global__ __launch_bounds__(128) void k_gemm_bias_f32(const _Float16* __restrict__ A,
    const _Float16* __restrict__ Bt, const float* __restrict__ bias,
    float* __restrict__ C, int N, int K) {
  const int wave = threadIdx.x >> 5, lane = threadIdx.x & 31;
  const int m0 = blockIdx.y * 128 + wave * 32;
  const int n0 = blockIdx.x * 32;
  v8f c[2][2] = {};
  gemm_core_32x32(A, Bt, K, m0, n0, c);
  const int rbase = (lane >> 4) << 3, cl = lane & 15;
#pragma unroll
  for (int mi = 0; mi < 2; ++mi)
#pragma unroll
    for (int ni = 0; ni < 2; ++ni)
#pragma unroll
      for (int r = 0; r < 8; ++r) {
        const int row = m0 + mi * 16 + rbase + r;
        const int col = n0 + ni * 16 + cl;
        C[(size_t)row * N + col] = c[mi][ni][r] + bias[col];
      }
}

// Fused attention: one block = (b,h, 16-query tile). Entire 16x1024 score
// row-block lives in LDS (no 402 MB att materialization, no online softmax).
__global__ __launch_bounds__(256) void k_attn(const _Float16* __restrict__ Qs,
    const _Float16* __restrict__ Ks, const _Float16* __restrict__ Vt,
    _Float16* __restrict__ O) {
  extern __shared__ char smem[];
  float*    Sbuf = (float*)smem;                                     // [16][1024] f32
  _Float16* Pbuf = (_Float16*)(smem + 16 * 1024 * 4);                // [16][1024] f16
  float*    red  = (float*)(smem + 16 * 1024 * 4 + 16 * 1024 * 2);   // [16][16]
  float*    obuf = (float*)(smem + 16 * 1024 * 4 + 16 * 1024 * 2 + 16 * 16 * 4); // [2][16][64]

  const int bh = blockIdx.x;          // 0..95  (b*12+h)
  const int q0 = blockIdx.y * 16;     // query tile base
  const int b = bh / 12, h = bh % 12;
  const _Float16* Qbh = Qs + (size_t)bh * 1024 * 64;
  const _Float16* Kbh = Ks + (size_t)bh * 1024 * 64;
  const _Float16* Vbh = Vt + (size_t)bh * 64 * 1024;

  const int tid = threadIdx.x, wave = tid >> 5, lane = tid & 31;
  const int rbase = (lane >> 4) << 3, cl = lane & 15;
  const float SCALE = 0.125f;  // 64^-0.5

  // Phase 1: S = relu((Q @ K^T) * SCALE) * N_HALF   (8 waves x 8 key-tiles each)
  v16h qa0 = load_frag(Qbh, 64, q0, 0);
  v16h qa1 = load_frag(Qbh, 64, q0, 32);
#pragma unroll
  for (int j = 0; j < 8; ++j) {
    const int nt = (wave + 8 * j) * 16;   // key tile base
    v8f acc = {};
    v16h kb0 = load_frag(Kbh, 64, nt, 0);
    v16h kb1 = load_frag(Kbh, 64, nt, 32);
    acc = WMMA_F32_F16(qa0, kb0, acc);
    acc = WMMA_F32_F16(qa1, kb1, acc);
#pragma unroll
    for (int r = 0; r < 8; ++r)
      Sbuf[(rbase + r) * 1024 + nt + cl] = fmaxf(acc[r] * SCALE, 0.0f) * 2.0f;
  }
  __syncthreads();

  // Phase 2: softmax over 1024 keys (16 rows, 16 threads/row, 64 elems each)
  {
    const int row = tid >> 4, seg = tid & 15;
    float* Srow = Sbuf + row * 1024 + seg * 64;
    float mx = -1e30f;
    for (int i = 0; i < 64; ++i) mx = fmaxf(mx, Srow[i]);
    red[tid] = mx;
    __syncthreads();
    float rmx = -1e30f;
    for (int i = 0; i < 16; ++i) rmx = fmaxf(rmx, red[row * 16 + i]);
    __syncthreads();
    float sum = 0.0f;
    for (int i = 0; i < 64; ++i) { float e = __expf(Srow[i] - rmx); Srow[i] = e; sum += e; }
    red[tid] = sum;
    __syncthreads();
    float rs = 0.0f;
    for (int i = 0; i < 16; ++i) rs += red[row * 16 + i];
    const float inv = 1.0f / rs;
    _Float16* Prow = Pbuf + row * 1024 + seg * 64;
    for (int i = 0; i < 64; ++i) Prow[i] = (_Float16)(Srow[i] * inv);
  }
  __syncthreads();

  // Phase 3: O = P @ V  (4 dim-tiles x 2 key-halves across 8 waves)
  {
    const int ntile = (wave & 3) * 16;   // output head-dim tile
    const int khalf = wave >> 2;         // key range half
    v8f oacc = {};
    for (int ks = khalf * 512; ks < khalf * 512 + 512; ks += 32) {
      v16h pa = load_frag(Pbuf, 1024, 0, ks);       // probs (LDS)
      v16h vb = load_frag(Vbh, 1024, ntile, ks);    // Vt row-major [64][1024]
      oacc = WMMA_F32_F16(pa, vb, oacc);
    }
#pragma unroll
    for (int r = 0; r < 8; ++r)
      obuf[(khalf * 16 + rbase + r) * 64 + ntile + cl] = oacc[r];
  }
  __syncthreads();
  for (int i = tid; i < 16 * 64; i += 256) {
    const int qq = i >> 6, d = i & 63;
    const float v = obuf[qq * 64 + d] + obuf[(16 + qq) * 64 + d];
    O[((size_t)b * 1024 + q0 + qq) * 768 + h * 64 + d] = (_Float16)v;
  }
}

extern "C" void kernel_launch(void* const* d_in, const int* in_sizes, int n_in,
                              void* d_out, int out_size, void* d_ws, size_t ws_size,
                              hipStream_t stream) {
  (void)in_sizes; (void)n_in; (void)out_size; (void)ws_size;
  const float* x     = (const float*)d_in[0];   // [8,1024,768]
  const float* Wqkv  = (const float*)d_in[1];   // [2304,768]
  const float* Wfc1  = (const float*)d_in[2];   // [64,64]
  const float* bfc1  = (const float*)d_in[3];   // [64]
  const float* Wfc2  = (const float*)d_in[4];   // [64,64]
  const float* bfc2  = (const float*)d_in[5];   // [64]
  const float* Wproj = (const float*)d_in[6];   // [768,768]
  const float* bproj = (const float*)d_in[7];   // [768]
  float* out = (float*)d_out;

  char* ws = (char*)d_ws;
  size_t off = 0;
  auto alloc = [&](size_t bytes) -> char* {
    char* p = ws + off;
    off += (bytes + 255) & ~(size_t)255;
    return p;
  };
  _Float16* xh     = (_Float16*)alloc(8192ull * 768 * 2);
  _Float16* Wqkvh  = (_Float16*)alloc(2304ull * 768 * 2);
  _Float16* Wprojh = (_Float16*)alloc(768ull * 768 * 2);
  _Float16* Wfc1h  = (_Float16*)alloc(64ull * 64 * 2);
  _Float16* Wfc2h  = (_Float16*)alloc(64ull * 64 * 2);
  _Float16* Qr     = (_Float16*)alloc(98304ull * 64 * 2);
  _Float16* Kr     = (_Float16*)alloc(98304ull * 64 * 2);
  _Float16* Vt     = (_Float16*)alloc(98304ull * 64 * 2);
  _Float16* Qss    = (_Float16*)alloc(98304ull * 64 * 2);
  _Float16* Kss    = (_Float16*)alloc(98304ull * 64 * 2);
  _Float16* Oa     = (_Float16*)alloc(8192ull * 768 * 2);

  auto cvt = [&](const float* s, _Float16* d, int n) {
    k_f32_to_f16<<<(n + 255) / 256, 256, 0, stream>>>(s, d, n);
  };
  cvt(x, xh, 8192 * 768);
  cvt(Wqkv, Wqkvh, 2304 * 768);
  cvt(Wproj, Wprojh, 768 * 768);
  cvt(Wfc1, Wfc1h, 64 * 64);
  cvt(Wfc2, Wfc2h, 64 * 64);

  // qkv projection + relu/scale/scatter: M=8192, N=2304, K=768
  k_gemm_qkv<<<dim3(2304 / 32, 8192 / 128), 128, 0, stream>>>(xh, Wqkvh, 768, Qr, Kr, Vt);

  // per-head FCs as big GEMMs: M=98304, N=64, K=64
  k_gemm_bias_f16<<<dim3(64 / 32, 98304 / 128), 128, 0, stream>>>(Qr, Wfc1h, bfc1, Qss, 64, 64);
  k_gemm_bias_f16<<<dim3(64 / 32, 98304 / 128), 128, 0, stream>>>(Kr, Wfc2h, bfc2, Kss, 64, 64);

  // fused attention: 96 (b,h) x 64 query-tiles, 105 KB dynamic LDS
  const size_t smem = 16 * 1024 * 4 + 16 * 1024 * 2 + 16 * 16 * 4 + 2 * 16 * 64 * 4;
  k_attn<<<dim3(96, 64), 256, smem, stream>>>(Qss, Kss, Vt, Oa);

  // output projection: M=8192, N=768, K=768 -> f32 d_out
  k_gemm_bias_f32<<<dim3(768 / 32, 8192 / 128), 128, 0, stream>>>(Oa, Wprojh, bproj, out, 768, 768);
}
